// LayerNormGRU_28999619183245
// MI455X (gfx1250) — matrix operand order repaired and checked by reading, not compiled, not run on hardware
//
#include <hip/hip_runtime.h>

#define B_   16
#define T_   2048
#define I_   1024
#define H_   1024
#define G_   3072
#define NWG_ 64
#define KP_  1032   // padded LDS row stride (elems) for the scan weight cache (TDM pad layout)
#define HP_  1040   // padded LDS row stride (elems) for staged A/h rows (16B-aligned)

typedef __attribute__((ext_vector_type(16))) __bf16 v16bf;
typedef __attribute__((ext_vector_type(8)))  float  v8f;
typedef __attribute__((ext_vector_type(4)))  unsigned v4u;
typedef __attribute__((ext_vector_type(8)))  int      v8i;
typedef __attribute__((ext_vector_type(4)))  int      v4i;

union ABu { v16bf v; unsigned int u[8]; uint4 q[2]; };

__device__ __forceinline__ unsigned short f2bf(float f) {
  unsigned int u = __float_as_uint(f);
  u += 0x7FFFu + ((u >> 16) & 1u);          // round-to-nearest-even
  return (unsigned short)(u >> 16);
}

__device__ __forceinline__ v8f wmma_bf16(v16bf a, v16bf b, v8f c) {
  return __builtin_amdgcn_wmma_f32_16x16x32_bf16(false, a, false, b, (short)0, c,
                                                 false, false);
}

// ---------------- CDNA5 async global->LDS copy (ASYNCcnt path) ----------------
__device__ __forceinline__ void async_copy_b128(void* lds_dst, const void* gsrc) {
  unsigned ldsa = (unsigned)(size_t)lds_dst;   // low 32 bits of flat LDS addr = LDS offset
  asm volatile("global_load_async_to_lds_b128 %0, %1, off"
               :: "v"(ldsa), "v"(gsrc)
               : "memory");
}
__device__ __forceinline__ void wait_async0() {
  asm volatile("s_wait_asynccnt 0" ::: "memory");
}

// ---------------- CDNA5 Tensor Data Mover: W slice -> LDS (TENSORcnt path) ----
// 3D tile: x = k (1024 bf16), y = 16 rows (stride 1024 elems), z = 3 gates
// (stride 1024*1024 elems). LDS padding: 2 DWORDs every 256 DWORDs stored ->
// effective row stride 1032 elems with a 4-elem gap after the first 512.
__device__ __forceinline__ void tdm_load_wslice(void* lds_base, const void* gbase) {
  unsigned lds_off = (unsigned)(size_t)lds_base;
  unsigned long long ga = (unsigned long long)(size_t)gbase;
  v4u g0 = { 1u,                                   // count=1, user descriptor
             lds_off,                              // lds_addr (bytes)
             (unsigned)ga,                         // global_addr[31:0]
             ((unsigned)(ga >> 32) & 0x01FFFFFFu) | 0x80000000u };  // addr[56:32] | type=2
  v8i g1 = { (int)0x03D10000,   // data_size=2B(code1)<<16 | pad_en<<20 | pad_int=7<<22 | pad_amt=1<<25
             (int)0x04000000,   // tensor_dim0[15:0]=1024 << 16
             (int)0x00100000,   // tensor_dim1[15:0]=16 << 16 (dim0 hi = 0)
             (int)0x04000000,   // tile_dim0=1024 << 16 (dim1 hi = 0)
             (int)0x00030010,   // tile_dim1=16 | tile_dim2=3 << 16
             1024,              // tensor_dim0_stride[31:0]
             0,                 // dim0_stride hi | dim1_stride[15:0] (1048576 & 0xffff = 0)
             16 };              // tensor_dim1_stride[47:16] = 1048576 >> 16
  v4i g2 = { 3, 0, 0, 0 };      // tensor_dim2=3; no iterate; dim2_stride unused (3D)
  v4i g3 = { 0, 0, 0, 0 };
#if __clang_major__ >= 23
  v8i g4 = { 0, 0, 0, 0, 0, 0, 0, 0 };
  __builtin_amdgcn_tensor_load_to_lds(g0, g1, g2, g3, g4, 0);
#else
  __builtin_amdgcn_tensor_load_to_lds(g0, g1, g2, g3, 0);
#endif
  __builtin_amdgcn_s_wait_tensorcnt((short)0);
}

// ---- WMMA tile loaders (16x16x32 bf16 shapes, wave32; A and B both use contiguous k = 16*(lane>>4) + s) ----

// A-matrix 16x32 bf16 from LDS rows of stride HP_ (staged by async copies).
__device__ __forceinline__ v16bf load_a_lds(const unsigned short* Al, int lane, int k0) {
  int m = lane & 15, half = lane >> 4;
  const unsigned short* p = Al + (size_t)m * HP_ + k0 + half * 16;
  ABu t;
  t.q[0] = *(const uint4*)(p);       // k0+half*16 .. +7
  t.q[1] = *(const uint4*)(p + 8);   // k0+half*16+8 .. +15
  return t.v;
}

// B-matrix 32x16 bf16 where B[k][n] = W[g0+n][k], W row-major [.. x ldk] bf16.
__device__ __forceinline__ v16bf load_b_bf16(const unsigned short* Wg, int lane, int ldk, int k0) {
  int n = lane & 15, kh = lane >> 4;
  const unsigned short* p = Wg + (size_t)n * ldk + k0 + kh * 16;
  ABu t;
  t.q[0] = *(const uint4*)(p);
  t.q[1] = *(const uint4*)(p + 8);
  return t.v;
}

// B-tile from the TDM-written LDS weight cache: row stride KP_=1032 elems with a
// 4-elem pad gap after the first 512 data elems of each row.
__device__ __forceinline__ v16bf load_b_lds(const unsigned short* wl, int gate, int lane, int k0) {
  int n = lane & 15, kh = lane >> 4;
  int k = k0 + kh * 16;
  k += (k >= 512) ? 4 : 0;
  const unsigned short* p = wl + (size_t)(gate * 16 + n) * KP_ + k;
  ABu t;
  t.q[0] = *(const uint4*)(p);
  t.q[1] = *(const uint4*)(p + 8);
  return t.v;
}

// C/D f32 16x16: VGPR j -> row (j + 8*(lane>>4)), col lane&15
__device__ __forceinline__ void store_c(float* p, int lane, int ldn, const v8f& c) {
  int n = lane & 15, mh = lane >> 4;
#pragma unroll
  for (int j = 0; j < 8; ++j) p[(size_t)(j + 8 * mh) * ldn + n] = c[j];
}

// -------------------- device-wide sense-reversing barrier --------------------
__device__ __forceinline__ void grid_barrier(int* bar, int nwg, int* s_sense) {
  __syncthreads();
  if (threadIdx.x == 0) {
    int next = 1 - *s_sense;
    *s_sense = next;
    __threadfence();
    int prev = __hip_atomic_fetch_add(&bar[0], 1, __ATOMIC_ACQ_REL, __HIP_MEMORY_SCOPE_AGENT);
    if (prev == nwg - 1) {
      __hip_atomic_store(&bar[0], 0, __ATOMIC_RELAXED, __HIP_MEMORY_SCOPE_AGENT);
      __hip_atomic_store(&bar[1], next, __ATOMIC_RELEASE, __HIP_MEMORY_SCOPE_AGENT);
    } else {
      while (__hip_atomic_load(&bar[1], __ATOMIC_ACQUIRE, __HIP_MEMORY_SCOPE_AGENT) != next)
        __builtin_amdgcn_s_sleep(2);
    }
  }
  __syncthreads();
}

// ------------------------------- kernels -------------------------------

// Convert weights + xs f32 -> bf16 (one bandwidth pass); reset barrier state.
__global__ void lngru_prep(const float* __restrict__ w_ih, const float* __restrict__ w_hh,
                           const float* __restrict__ xs,
                           unsigned short* __restrict__ wih_bf, unsigned short* __restrict__ whh_bf,
                           unsigned short* __restrict__ xs_bf,
                           int* bar) {
  int tid = blockIdx.x * blockDim.x + threadIdx.x;
  int stride = gridDim.x * blockDim.x;
  if (tid == 0) { bar[0] = 0; bar[1] = 0; }
  const int NW = G_ * I_;
  for (int i = tid; i < NW; i += stride) {
    wih_bf[i] = f2bf(w_ih[i]);
    whh_bf[i] = f2bf(w_hh[i]);
  }
  const int NX = B_ * T_ * I_;
  for (int i = tid; i < NX; i += stride) xs_bf[i] = f2bf(xs[i]);
}

// proj[bt, g] = sum_i xs[bt,i] * w_ih[g,i]   (bf16 WMMA, f32 accumulate)
// Block = one M-tile (16 rows of bt) x 512 columns. A staged to LDS via async DMA,
// shared by 8 waves; each wave owns 4 N-tiles (A-fragment reuse x4).
__global__ void lngru_gemm_ih(const unsigned short* __restrict__ xs_bf,
                              const unsigned short* __restrict__ wih_bf,
                              float* __restrict__ proj) {
  __shared__ unsigned short a_lds[16 * HP_];     // 32.5 KB
  const int tid = threadIdx.x;
  const int wave = tid >> 5, lane = tid & 31;
  const int mt = blockIdx.x / 6;                 // 2048 M-tiles
  const int ng = blockIdx.x % 6;                 // 6 column groups of 512

  // Stage A tile (16 x 1024 bf16) into LDS with async copies: 2048 16B chunks.
  const unsigned short* A = xs_bf + (size_t)(mt * 16) * I_;
#pragma unroll
  for (int i = 0; i < 8; ++i) {
    int c = tid + i * 256;                       // 0..2047
    int m = c >> 7, off = (c & 127) * 8;
    async_copy_b128(a_lds + (size_t)m * HP_ + off, A + (size_t)m * I_ + off);
  }
  wait_async0();
  __syncthreads();

  const int nt0 = ng * 32 + wave * 4;            // first of this wave's 4 N-tiles
  const unsigned short* B0 = wih_bf + (size_t)(nt0 * 16) * I_;
  v8f acc0 = {}, acc1 = {}, acc2 = {}, acc3 = {};
#pragma unroll 2
  for (int k0 = 0; k0 < I_; k0 += 32) {
    v16bf a = load_a_lds(a_lds, lane, k0);
    acc0 = wmma_bf16(a, load_b_bf16(B0,                        lane, I_, k0), acc0);
    acc1 = wmma_bf16(a, load_b_bf16(B0 + (size_t)16 * I_,      lane, I_, k0), acc1);
    acc2 = wmma_bf16(a, load_b_bf16(B0 + (size_t)32 * I_,      lane, I_, k0), acc2);
    acc3 = wmma_bf16(a, load_b_bf16(B0 + (size_t)48 * I_,      lane, I_, k0), acc3);
  }
  float* C = proj + (size_t)(mt * 16) * G_ + (size_t)nt0 * 16;
  store_c(C,      lane, G_, acc0);
  store_c(C + 16, lane, G_, acc1);
  store_c(C + 32, lane, G_, acc2);
  store_c(C + 48, lane, G_, acc3);
}

__device__ __forceinline__ float block_reduce(float v, float* red) {
  int tid = threadIdx.x;
  red[tid] = v; __syncthreads();
  for (int o = 128; o > 0; o >>= 1) {
    if (tid < o) red[tid] += red[tid + o];
    __syncthreads();
  }
  float r = red[0]; __syncthreads();
  return r;
}

// In-place LayerNorm of proj rows: cols [0,2048) (var /2047), cols [2048,3072) (var /1023).
__global__ void lngru_ln_ih(float* __restrict__ proj,
                            const float* __restrict__ gamma,
                            const float* __restrict__ beta) {
  __shared__ float red[256];
  float* p = proj + (size_t)blockIdx.x * G_;
  int tid = threadIdx.x;

  float x1[8], s1 = 0.f, s2 = 0.f;
#pragma unroll
  for (int u = 0; u < 8; ++u) { float v = p[tid + u * 256]; x1[u] = v; s1 += v; s2 += v * v; }
  float sum   = block_reduce(s1, red);
  float sumsq = block_reduce(s2, red);
  float mean = sum * (1.0f / 2048.0f);
  float var  = (sumsq - 2048.0f * mean * mean) * (1.0f / 2047.0f);
  float rstd = rsqrtf(var);
#pragma unroll
  for (int u = 0; u < 8; ++u) {
    int c = tid + u * 256;
    p[c] = gamma[c] * (x1[u] - mean) * rstd + beta[c];
  }

  float x2[4]; s1 = 0.f; s2 = 0.f;
#pragma unroll
  for (int u = 0; u < 4; ++u) { float v = p[2048 + tid + u * 256]; x2[u] = v; s1 += v; s2 += v * v; }
  sum   = block_reduce(s1, red);
  sumsq = block_reduce(s2, red);
  mean = sum * (1.0f / 1024.0f);
  var  = (sumsq - 1024.0f * mean * mean) * (1.0f / 1023.0f);
  rstd = rsqrtf(var);
#pragma unroll
  for (int u = 0; u < 4; ++u) {
    int c = 2048 + tid + u * 256;
    p[c] = gamma[c] * (x2[u] - mean) * rstd + beta[c];
  }
}

// Persistent recurrent scan: 64 WGs x 256 threads. WG owns 16 hidden columns;
// its 3 weight tiles (~99KB bf16) are DMA'd into LDS once by the Tensor Data
// Mover; h (32KB bf16) is staged into LDS each step via the async-to-LDS path.
__global__ void lngru_scan(const float* __restrict__ proj,
                           const float* __restrict__ h0,
                           const unsigned short* __restrict__ whh_bf,
                           const float* __restrict__ b_hh,
                           const float* __restrict__ g_hh,
                           unsigned short* __restrict__ h_bf,
                           float* __restrict__ part_rz,
                           float* __restrict__ part_n,
                           int* bar,
                           float* __restrict__ ys,
                           float* __restrict__ hfin) {
  __shared__ unsigned short wl[48 * KP_];        // TDM-filled weight cache (~99KB)
  __shared__ unsigned short h_lds[16 * HP_];     // staged h (~32.5KB)
  __shared__ float hhred[3 * 8 * 256];           // per-wave partial C tiles (24KB)
  __shared__ float s_rz[256], s_nn[256];
  __shared__ float s_mrz[16], s_mn[16];
  __shared__ int   s_sense;

  const int tid = threadIdx.x, wg = blockIdx.x;
  const int wave = tid >> 5, lane = tid & 31;
  const int col = tid & 15, m = tid >> 4;        // epilogue element (batch m, col)
  const int j0 = wg * 16, j = j0 + col;

  if (tid == 0) s_sense = 0;

  // One TDM descriptor moves the whole 3-gate weight slice into padded LDS rows.
  if (wave == 0) tdm_load_wslice(wl, whh_bf + (size_t)j0 * H_);

  // Each thread owns h[m][j] in a register for the whole scan.
  float hreg = h0[m * H_ + j];
  h_bf[m * H_ + j] = f2bf(hreg);
  const float gamv[3] = { g_hh[j], g_hh[H_ + j], g_hh[2 * H_ + j] };
  const float betv[3] = { b_hh[j], b_hh[H_ + j], b_hh[2 * H_ + j] };

  grid_barrier(bar, NWG_, &s_sense);             // h_bf + LDS caches ready everywhere

  for (int t = 0; t < T_; ++t) {
    size_t rowoff = ((size_t)m * T_ + t) * G_;
    __builtin_prefetch(&proj[rowoff + j], 0, 1); // warm this step's x while GEMM runs

    // Stage h (16 x 1024 bf16) into LDS: 2048 x 16B async chunks.
#pragma unroll
    for (int i = 0; i < 8; ++i) {
      int c = tid + i * 256;
      int m2 = c >> 7, off = (c & 127) * 8;
      async_copy_b128(h_lds + (size_t)m2 * HP_ + off, h_bf + (size_t)m2 * H_ + off);
    }
    wait_async0();
    __syncthreads();

    // hh = h @ W^T : wave w covers K in [w*128, w*128+128) for all 3 gate tiles.
    v8f acc0 = {}, acc1 = {}, acc2 = {};
#pragma unroll
    for (int i = 0; i < 4; ++i) {
      int k0 = wave * 128 + i * 32;
      v16bf a = load_a_lds(h_lds, lane, k0);
      acc0 = wmma_bf16(a, load_b_lds(wl, 0, lane, k0), acc0);
      acc1 = wmma_bf16(a, load_b_lds(wl, 1, lane, k0), acc1);
      acc2 = wmma_bf16(a, load_b_lds(wl, 2, lane, k0), acc2);
    }
    {
      int nn = lane & 15, mh = lane >> 4;
#pragma unroll
      for (int q = 0; q < 8; ++q) {
        int e = (q + 8 * mh) * 16 + nn;
        hhred[0 * 2048 + wave * 256 + e] = acc0[q];
        hhred[1 * 2048 + wave * 256 + e] = acc1[q];
        hhred[2 * 2048 + wave * 256 + e] = acc2[q];
      }
    }
    __syncthreads();

    float hr = 0.f, hz = 0.f, hn = 0.f;
#pragma unroll
    for (int w = 0; w < 8; ++w) {
      hr += hhred[0 * 2048 + w * 256 + tid];
      hz += hhred[1 * 2048 + w * 256 + tid];
      hn += hhred[2 * 2048 + w * 256 + tid];
    }
    s_rz[tid] = hr + hz;
    s_nn[tid] = hn;
    __syncthreads();

    if (tid < 16) {                              // per-batch partial sums over our 16 cols
      float a = 0.f, b = 0.f;
#pragma unroll
      for (int c = 0; c < 16; ++c) { a += s_rz[tid * 16 + c]; b += s_nn[tid * 16 + c]; }
      part_rz[wg * 16 + tid] = a;
      part_n [wg * 16 + tid] = b;
    }
    grid_barrier(bar, NWG_, &s_sense);           // partial means published

    if (tid < 16) {
      float a = 0.f, b = 0.f;
      for (int w = 0; w < NWG_; ++w) { a += part_rz[w * 16 + tid]; b += part_n[w * 16 + tid]; }
      s_mrz[tid] = a * (1.0f / 2048.0f);
      s_mn [tid] = b * (1.0f / 1024.0f);
    }
    __syncthreads();

    float mrz = s_mrz[m], mn = s_mn[m];
    float xr = proj[rowoff + j];
    float xz = proj[rowoff + H_ + j];
    float xn = proj[rowoff + 2 * H_ + j];
    float lr = gamv[0] * (hr - mrz) + betv[0];
    float lz = gamv[1] * (hz - mrz) + betv[1];
    float ln = gamv[2] * (hn - mn ) + betv[2];
    float r  = 1.0f / (1.0f + expf(-(xr + lr)));
    float z  = 1.0f / (1.0f + expf(-(xz + lz)));
    float nv = tanhf(xn + r * ln);
    hreg = (1.0f - z) * nv + z * hreg;

    ys[((size_t)m * T_ + t) * H_ + j] = hreg;
    h_bf[m * H_ + j] = f2bf(hreg);

    grid_barrier(bar, NWG_, &s_sense);           // h visible before next step's GEMM
  }

  hfin[m * H_ + j] = hreg;
}

// ------------------------------- launcher -------------------------------
extern "C" void kernel_launch(void* const* d_in, const int* in_sizes, int n_in,
                              void* d_out, int out_size, void* d_ws, size_t ws_size,
                              hipStream_t stream) {
  (void)in_sizes; (void)n_in; (void)out_size; (void)ws_size;
  const float* xs   = (const float*)d_in[0];
  const float* h0   = (const float*)d_in[1];
  const float* w_ih = (const float*)d_in[2];
  const float* w_hh = (const float*)d_in[3];
  const float* b_ih = (const float*)d_in[4];
  const float* b_hh = (const float*)d_in[5];
  const float* g_ih = (const float*)d_in[6];
  const float* g_hh = (const float*)d_in[7];

  float* out  = (float*)d_out;
  float* ys   = out;                                   // (B,T,H)
  float* hfin = out + (size_t)B_ * T_ * H_;            // (B,H)

  char* ws = (char*)d_ws;
  const size_t SZ_PROJ = (size_t)B_ * T_ * G_ * 4;     // 402,653,184
  const size_t SZ_W    = (size_t)G_ * I_ * 2;          // 6,291,456
  const size_t SZ_XSBF = (size_t)B_ * T_ * I_ * 2;     // 67,108,864
  float*          proj    = (float*)ws;
  unsigned short* wih_bf  = (unsigned short*)(ws + SZ_PROJ);
  unsigned short* whh_bf  = (unsigned short*)(ws + SZ_PROJ + SZ_W);
  unsigned short* xs_bf   = (unsigned short*)(ws + SZ_PROJ + 2 * SZ_W);
  unsigned short* h_bf    = (unsigned short*)(ws + SZ_PROJ + 2 * SZ_W + SZ_XSBF);
  float*          part_rz = (float*)(ws + SZ_PROJ + 2 * SZ_W + SZ_XSBF + 32768);
  float*          part_n  = (float*)(ws + SZ_PROJ + 2 * SZ_W + SZ_XSBF + 32768 + 4096);
  int*            bar     = (int*)  (ws + SZ_PROJ + 2 * SZ_W + SZ_XSBF + 32768 + 8192);

  lngru_prep<<<2048, 256, 0, stream>>>(w_ih, w_hh, xs, wih_bf, whh_bf, xs_bf, bar);

  // 2048 M-tiles x 6 column groups (512 cols each)
  lngru_gemm_ih<<<2048 * 6, 256, 0, stream>>>(xs_bf, wih_bf, proj);

  lngru_ln_ih<<<B_ * T_, 256, 0, stream>>>(proj, g_ih, b_ih);

  lngru_scan<<<NWG_, 256, 0, stream>>>(proj, h0, whh_bf, b_hh, g_hh, h_bf,
                                       part_rz, part_n, bar, ys, hfin);
}
